// PredictingModule_88622355186221
// MI455X (gfx1250) — matrix operand order repaired and checked by reading, not compiled, not run on hardware
//
#include <hip/hip_runtime.h>

typedef __attribute__((ext_vector_type(16))) _Float16 v16h;
typedef __attribute__((ext_vector_type(8)))  float    v8f;
typedef __attribute__((ext_vector_type(2)))  _Float16 h2;
typedef __attribute__((ext_vector_type(4)))  unsigned int u32x4;
typedef __attribute__((ext_vector_type(4)))  int      i32x4;
typedef __attribute__((ext_vector_type(8)))  int      i32x8;

#define EMB_D      128
#define H1         128
#define W1_TILES   64                    // 8 n-tiles x 8 k-steps
#define W2_TILES   8                     // 2 n-tiles x 4 k-steps
#define PACK_DWORDS ((W1_TILES + W2_TILES) * 256)   // 18432 dwords = 72 KB
#define MISC_FLOATS 193                  // b1[128] b2[32] w3[32] b3[1]
#define X2_STRIDE   136                  // halfs per x2 row (272 B, breaks bank conflicts)

// ---------------------------------------------------------------------------
// Pack w1/w2 (fp32, row-major [out,in]) into f16 WMMA B-operand register
// images: per 16x16x32 tile, dword = lane*8 + v ; lane<16 covers K=2v+8*(v>=4),+1
// (K 0..7,16..23), lane>=16 covers those K+8 (K 8..15,24..31); column N = lane%16.
// Layer-2 K dimension is PERMUTED: storage col c <-> hidden (c&7)*16 + (c>>3),
// matching the vectorized layer-1 -> LDS transpose in the main kernel.
// ---------------------------------------------------------------------------
__global__ void pack_weights_kernel(const float* __restrict__ w1,
                                    const float* __restrict__ b1,
                                    const float* __restrict__ w2,
                                    const float* __restrict__ b2,
                                    const float* __restrict__ w3,
                                    const float* __restrict__ b3,
                                    unsigned int* __restrict__ ws)
{
    int tid = blockIdx.x * blockDim.x + threadIdx.x;
    if (tid < W1_TILES * 256) {
        int t = tid >> 8, r = tid & 255;
        int l = r >> 3, v = r & 7;
        int ntile = t >> 3, s = t & 7;                 // tile = ntile*8 + kstep
        int n = ntile * 16 + (l & 15);                 // output unit (B column)
        int k = s * 32 + ((l < 16) ? 0 : 8) + 2 * v + ((v >= 4) ? 8 : 0);  // input dim (B row)
        union { h2 h; unsigned int u; } p;
        p.h[0] = (_Float16)w1[n * 256 + k];
        p.h[1] = (_Float16)w1[n * 256 + k + 1];
        ws[t * 256 + l * 8 + v] = p.u;
    } else if (tid < PACK_DWORDS) {
        int u = tid - W1_TILES * 256;
        int t = u >> 8, r = u & 255;
        int l = r >> 3, v = r & 7;
        int ntile = t >> 2, s = t & 3;                 // tile = ntile*4 + kstep
        int n = ntile * 16 + (l & 15);
        int kp = s * 32 + ((l < 16) ? 0 : 8) + 2 * v + ((v >= 4) ? 8 : 0); // storage column (K')
        int h0 = ((kp & 7) << 4) | (kp >> 3);          // actual hidden index
        int h1 = (((kp + 1) & 7) << 4) | ((kp + 1) >> 3);
        union { h2 h; unsigned int u; } p;
        p.h[0] = (_Float16)w2[n * 128 + h0];
        p.h[1] = (_Float16)w2[n * 128 + h1];
        ws[(W1_TILES + t) * 256 + l * 8 + v] = p.u;
    } else if (tid < PACK_DWORDS + MISC_FLOATS) {
        int i = tid - PACK_DWORDS;
        float val = (i < 128) ? b1[i]
                  : (i < 160) ? b2[i - 128]
                  : (i < 192) ? w3[i - 160]
                  : b3[0];
        ((float*)ws)[PACK_DWORDS + i] = val;
    }
}

// Build f16 A-operand (16x32 tile) from a row of fp32 features.
__device__ __forceinline__ v16h load_cvt_A(const float* base)
{
    float4 f0 = *(const float4*)(base);
    float4 f1 = *(const float4*)(base + 4);
    float4 f2 = *(const float4*)(base + 16);
    float4 f3 = *(const float4*)(base + 20);
    v16h a;
    a[0]  = (_Float16)f0.x; a[1]  = (_Float16)f0.y; a[2]  = (_Float16)f0.z; a[3]  = (_Float16)f0.w;
    a[4]  = (_Float16)f1.x; a[5]  = (_Float16)f1.y; a[6]  = (_Float16)f1.z; a[7]  = (_Float16)f1.w;
    a[8]  = (_Float16)f2.x; a[9]  = (_Float16)f2.y; a[10] = (_Float16)f2.z; a[11] = (_Float16)f2.w;
    a[12] = (_Float16)f3.x; a[13] = (_Float16)f3.y; a[14] = (_Float16)f3.z; a[15] = (_Float16)f3.w;
    return a;
}

__device__ __forceinline__ float sigmoidf(float x)
{
    return 1.0f / (1.0f + __expf(-x));
}

// Single LDS object => base offset 0, so member offsets are the LDS addresses.
// Weight block deliberately LAST: any TDM-descriptor overflow falls off the
// allocation instead of clobbering live data.
struct SMem {
    _Float16 x2[8][16 * X2_STRIDE];   // 34816 B : per-wave transpose buffers
    float    misc[196];               //   784 B : biases + w3 (+pad)
    uint4    w[PACK_DWORDS / 4];      // 73728 B : packed W1+W2 (TDM target)
};
#define W_LDS_OFF 35600
static_assert(__builtin_offsetof(SMem, w) == W_LDS_OFF, "LDS layout");

union BU { uint4 q[2]; v16h h; };

// ---------------------------------------------------------------------------
// Main kernel: 8 waves/WG; weights LDS-resident (TDM-staged); each wave
// processes 32-edge units (2 WMMA M-tiles sharing every B-tile load).
// ---------------------------------------------------------------------------
__global__ __launch_bounds__(256) void mlp_edge_kernel(
    const float* __restrict__ h_customer,
    const float* __restrict__ h_product,
    const int*   __restrict__ src_cp, const int* __restrict__ dst_cp,
    const int*   __restrict__ src_pc, const int* __restrict__ dst_pc,
    const unsigned int* __restrict__ ws,
    float* __restrict__ out, int E)
{
    __shared__ SMem sm;

    const int lane = threadIdx.x & 31;
    const int wave = threadIdx.x >> 5;
    const int hi   = lane >> 4;
    const int lm   = lane & 15;

    // ---- stage packed weights into LDS via the Tensor Data Mover ----------
    if (wave == 0) {
        unsigned long long ga = (unsigned long long)(uintptr_t)ws;
        u32x4 g0 = {};
        g0[0] = 1u;                                   // count = 1 descriptor
        g0[1] = (unsigned)W_LDS_OFF;                  // lds_addr
        g0[2] = (unsigned)ga;                         // global_addr[31:0]
        g0[3] = (unsigned)((ga >> 32) & 0x01FFFFFFu)  // global_addr[56:32]
              | 0x80000000u;                          // type = 2 ("image")
        i32x8 g1 = {};
        g1[0] = 0x00030000;                           // data_size=3 (8B), no multicast
        g1[1] = (int)(9216u << 16);                   // tensor_dim0 = 9216 elems (72KB)
        g1[2] = 0;
        g1[3] = (int)(9216u << 16);                   // tile_dim0 = 9216 (1-D copy)
        g1[5] = 9216;                                 // tensor_dim0_stride
        i32x4 z4 = {};
        i32x8 z8 = {};
        __builtin_amdgcn_tensor_load_to_lds(g0, g1, z4, z4, z8, 0);
        __builtin_amdgcn_s_wait_tensorcnt(0);
    }
    // Compile-only environment: TDM semantics unverifiable, so a cheap
    // vector copy over the same region guarantees the weight contents.
    for (int i = threadIdx.x; i < PACK_DWORDS / 4; i += blockDim.x)
        sm.w[i] = ((const uint4*)ws)[i];
    for (int i = threadIdx.x; i < MISC_FLOATS; i += blockDim.x)
        sm.misc[i] = ((const float*)ws)[PACK_DWORDS + i];
    __syncthreads();

    const int unitsPerType = E / 32;
    const int totalUnits   = unitsPerType * 2;

    _Float16* x2 = sm.x2[wave];

    float bias1[8];
    #pragma unroll
    for (int n = 0; n < 8; ++n) bias1[n] = sm.misc[n * 16 + lm];

    for (int u = blockIdx.x * 8 + wave; u < totalUnits; u += gridDim.x * 8) {
        const int etype = (u >= unitsPerType);
        const int eb    = (etype ? (u - unitsPerType) : u) * 32;
        const int*   src = etype ? src_pc : src_cp;
        const int*   dst = etype ? dst_pc : dst_cp;
        const float* hs  = etype ? h_product  : h_customer;  // dims 0..127
        const float* hd  = etype ? h_customer : h_product;   // dims 128..255
        float* ob = out + (size_t)etype * (size_t)E + eb;

        const float* pS[2]; const float* pD[2];
        #pragma unroll
        for (int m = 0; m < 2; ++m) {
            int e = eb + m * 16 + lm;                // row M = lane%16
            pS[m] = hs + (size_t)src[e] * EMB_D + hi * 8;
            pD[m] = hd + (size_t)dst[e] * EMB_D + hi * 8;
        }

        // prefetch next unit's gathered rows (one row per lane)
        {
            int un = u + gridDim.x * 8;
            if (un < totalUnits) {
                int et2 = (un >= unitsPerType);
                int eb2 = (et2 ? (un - unitsPerType) : un) * 32;
                const int* sp2 = et2 ? src_pc : src_cp;
                const int* dp2 = et2 ? dst_pc : dst_cp;
                const float* hs2 = et2 ? h_product  : h_customer;
                const float* hd2 = et2 ? h_customer : h_product;
                const float* ps2 = hs2 + (size_t)sp2[eb2 + lane] * EMB_D;
                const float* pd2 = hd2 + (size_t)dp2[eb2 + lane] * EMB_D;
                __builtin_prefetch(ps2, 0, 1);
                __builtin_prefetch(ps2 + 64, 0, 1);
                __builtin_prefetch(pd2, 0, 1);
                __builtin_prefetch(pd2 + 64, 0, 1);
            }
        }

        // ---------------- layer 1: [32 x 256] x [256 x 128] ----------------
        v8f acc[2][8];
        #pragma unroll
        for (int m = 0; m < 2; ++m)
            #pragma unroll
            for (int n = 0; n < 8; ++n)
                acc[m][n] = (v8f){};

        #pragma unroll
        for (int s = 0; s < 8; ++s) {
            v16h A0 = load_cvt_A((s < 4) ? (pS[0] + s * 32) : (pD[0] + (s - 4) * 32));
            v16h A1 = load_cvt_A((s < 4) ? (pS[1] + s * 32) : (pD[1] + (s - 4) * 32));
            const uint4* bp = &sm.w[s * 64 + lane * 2];   // tile (n=0, kstep=s)
            BU Bc;
            Bc.q[0] = bp[0]; Bc.q[1] = bp[1];
            #pragma unroll
            for (int n = 0; n < 8; ++n) {
                BU Bn;                                    // double buffer: next tile
                if (n < 7) {
                    const uint4* np = bp + (n + 1) * 512; // +8 tiles * 64 uint4
                    Bn.q[0] = np[0]; Bn.q[1] = np[1];
                }
                acc[0][n] = __builtin_amdgcn_wmma_f32_16x16x32_f16(
                    false, A0, false, Bc.h, (short)0, acc[0][n], false, false);
                acc[1][n] = __builtin_amdgcn_wmma_f32_16x16x32_f16(
                    false, A1, false, Bc.h, (short)0, acc[1][n], false, false);
                if (n < 7) Bc = Bn;
            }
        }

        // ------------- layers 2+3 per M-tile (16 edges each) ---------------
        #pragma unroll
        for (int m = 0; m < 2; ++m) {
            // bias + ReLU; vectorized transpose store: row = edge, permuted
            // column c = lm*8 + n  (one b128 store per r)
            #pragma unroll
            for (int r = 0; r < 8; ++r) {
                union { _Float16 h[8]; uint4 q; } pk;
                #pragma unroll
                for (int n = 0; n < 8; ++n) {
                    float v = acc[m][n][r] + bias1[n];
                    v = v > 0.0f ? v : 0.0f;
                    pk.h[n] = (_Float16)v;
                }
                *(uint4*)&x2[(r + hi * 8) * X2_STRIDE + lm * 8] = pk.q;
            }
            // per-wave LDS ops are in-order; fence compiler + HW
            asm volatile("s_wait_dscnt 0x0" ::: "memory");

            // layer 2: [16 x 128] x [128 x 32] over permuted K'
            v8f acc2[2]; acc2[0] = (v8f){}; acc2[1] = (v8f){};
            #pragma unroll
            for (int s2 = 0; s2 < 4; ++s2) {
                const _Float16* xr = x2 + lm * X2_STRIDE + s2 * 32 + hi * 8;
                BU A2;
                A2.q[0] = *(const uint4*)(xr);
                A2.q[1] = *(const uint4*)(xr + 16);
                #pragma unroll
                for (int n2 = 0; n2 < 2; ++n2) {
                    BU B;
                    const uint4* bp = &sm.w[(W1_TILES + n2 * 4 + s2) * 64 + lane * 2];
                    B.q[0] = bp[0]; B.q[1] = bp[1];
                    acc2[n2] = __builtin_amdgcn_wmma_f32_16x16x32_f16(
                        false, A2.h, false, B.h, (short)0, acc2[n2], false, false);
                }
            }
            asm volatile("s_wait_dscnt 0x0" ::: "memory");

            // layer 3: bias + ReLU + dot(w3) + 16-lane reduce + sigmoid
            float sum[8];
            #pragma unroll
            for (int r = 0; r < 8; ++r) sum[r] = 0.0f;
            #pragma unroll
            for (int n2 = 0; n2 < 2; ++n2) {
                int   k   = n2 * 16 + lm;
                float b2v = sm.misc[128 + k];
                float w3v = sm.misc[160 + k];
                #pragma unroll
                for (int r = 0; r < 8; ++r) {
                    float y = acc2[n2][r] + b2v;
                    y = y > 0.0f ? y : 0.0f;
                    sum[r] += y * w3v;
                }
            }
            #pragma unroll
            for (int r = 0; r < 8; ++r) {
                sum[r] += __shfl_xor(sum[r], 1, 16);
                sum[r] += __shfl_xor(sum[r], 2, 16);
                sum[r] += __shfl_xor(sum[r], 4, 16);
                sum[r] += __shfl_xor(sum[r], 8, 16);
            }
            if (lm == 0) {
                float b3v = sm.misc[192];
                float4 o0, o1;
                o0.x = sigmoidf(sum[0] + b3v); o0.y = sigmoidf(sum[1] + b3v);
                o0.z = sigmoidf(sum[2] + b3v); o0.w = sigmoidf(sum[3] + b3v);
                o1.x = sigmoidf(sum[4] + b3v); o1.y = sigmoidf(sum[5] + b3v);
                o1.z = sigmoidf(sum[6] + b3v); o1.w = sigmoidf(sum[7] + b3v);
                float* p = ob + m * 16 + hi * 8;
                *(float4*)(p)     = o0;
                *(float4*)(p + 4) = o1;
            }
        }
    }
}

extern "C" void kernel_launch(void* const* d_in, const int* in_sizes, int n_in,
                              void* d_out, int out_size, void* d_ws, size_t ws_size,
                              hipStream_t stream)
{
    const float* h_customer = (const float*)d_in[0];
    const float* h_product  = (const float*)d_in[1];
    const int*   src_cp     = (const int*)d_in[2];
    const int*   dst_cp     = (const int*)d_in[3];
    const int*   src_pc     = (const int*)d_in[4];
    const int*   dst_pc     = (const int*)d_in[5];
    const float* w1 = (const float*)d_in[6];
    const float* b1 = (const float*)d_in[7];
    const float* w2 = (const float*)d_in[8];
    const float* b2 = (const float*)d_in[9];
    const float* w3 = (const float*)d_in[10];
    const float* b3 = (const float*)d_in[11];
    const int E = in_sizes[2];

    unsigned int* ws = (unsigned int*)d_ws;

    int packThreads = PACK_DWORDS + MISC_FLOATS;
    hipLaunchKernelGGL(pack_weights_kernel,
                       dim3((packThreads + 255) / 256), dim3(256), 0, stream,
                       w1, b1, w2, b2, w3, b3, ws);

    int units  = (E / 32) * 2;
    int blocks = (units + 7) / 8;
    if (blocks > 1024) blocks = 1024;
    hipLaunchKernelGGL(mlp_edge_kernel, dim3(blocks), dim3(256), 0, stream,
                       h_customer, h_product, src_cp, dst_cp, src_pc, dst_pc,
                       ws, (float*)d_out, E);
}
